// Tomer_61856118997013
// MI455X (gfx1250) — compile-verified, matrix operand order checked by reading, not run on hardware
//
#include <hip/hip_runtime.h>
#include <hip/hip_fp16.h>
#include <cstdint>

// ---------------------------------------------------------------------------
// Point-cloud transformer forward (ToMe + KNN attention) for MI455X / gfx1250.
// Dense GEMMs: v_wmma_f32_16x16x32_f16, f16 operands staged through LDS with
// global_load_async_to_lds_b128 (ASYNCcnt) when available.
// ---------------------------------------------------------------------------

#define EMBED   256
#define HEADS   8
#define HD      32
#define KNN_K   16
#define DEPTH   4
#define TOME_R  512
#define LASTF   16
#define MLP_HID 2048

typedef __attribute__((ext_vector_type(16))) _Float16 v16h;
typedef __attribute__((ext_vector_type(8)))  _Float16 v8h;
typedef __attribute__((ext_vector_type(8)))  float    v8f;
typedef __attribute__((ext_vector_type(4)))  int      v4i;

#define GLOBAL_AS __attribute__((address_space(1)))
#define LDS_AS    __attribute__((address_space(3)))
typedef GLOBAL_AS v4i* gv4i_p;
typedef LDS_AS    v4i* lv4i_p;

#if defined(__has_builtin)
#if __has_builtin(__builtin_amdgcn_global_load_async_to_lds_b128) && \
    __has_builtin(__builtin_amdgcn_s_wait_asynccnt)
#define HAVE_ASYNC_LDS 1
#endif
#endif
#ifndef HAVE_ASYNC_LDS
#define HAVE_ASYNC_LDS 0
#endif

// ---------------------------------------------------------------------------
// WMMA GEMM: C[M,N] = A[M,K] * Bt^T + bias, A:[M][K] f16, Bt:[N][K] f16.
// M,N multiples of 64; K multiple of 32 (all call sites satisfy this).
// Block = 128 threads = 4 waves; tile 64x64; wave = 16 rows x four 16x16 WMMA.
// ---------------------------------------------------------------------------
#define TM  64
#define TN  64
#define TK  32
#define LDK 40   // padded K stride in halves (80B rows, 16B aligned)

__global__ __launch_bounds__(128) void k_gemm_wmma(
    const _Float16* __restrict__ A, const _Float16* __restrict__ Bt,
    const float* __restrict__ bias, float* __restrict__ C,
    int M, int N, int K)
{
  (void)M;
  __shared__ __align__(16) _Float16 Ash[TM * LDK];
  __shared__ __align__(16) _Float16 Bsh[TN * LDK];
  const int tid  = threadIdx.x;
  const int lane = tid & 31;
  const int wv   = tid >> 5;        // wave 0..3 -> M sub-tile
  const int bm   = blockIdx.y * TM;
  const int bn   = blockIdx.x * TN;
  const int r    = lane & 15;       // M row (A) / N col (B) within tile
  const int hi   = lane >> 4;       // K-half selector per ISA layout

  v8f acc[4] = {};

  for (int k0 = 0; k0 < K; k0 += TK) {
    if (k0 + TK < K) {   // pull next K tile toward the caches (global_prefetch_b8)
      __builtin_prefetch(&A[(size_t)(bm + (tid >> 1)) * K + k0 + TK], 0, 1);
      __builtin_prefetch(&Bt[(size_t)(bn + (tid >> 1)) * K + k0 + TK], 0, 1);
    }
    // Tile = 64 rows x 32 halves = 256 x 16B chunks; 128 threads -> 2 chunks each.
#pragma unroll
    for (int t = 0; t < 2; ++t) {
      int c   = t * 128 + tid;
      int row = c >> 2;             // 0..63
      int c8  = c & 3;              // 8-half group within the row
      const v8h* gA = (const v8h*)&A[(size_t)(bm + row) * K + k0];
      const v8h* gB = (const v8h*)&Bt[(size_t)(bn + row) * K + k0];
      _Float16* lA = &Ash[row * LDK + c8 * 8];
      _Float16* lB = &Bsh[row * LDK + c8 * 8];
#if HAVE_ASYNC_LDS
      __builtin_amdgcn_global_load_async_to_lds_b128(
          (gv4i_p)(uintptr_t)(gA + c8),
          (lv4i_p)(unsigned)(uintptr_t)lA, 0, 0);
      __builtin_amdgcn_global_load_async_to_lds_b128(
          (gv4i_p)(uintptr_t)(gB + c8),
          (lv4i_p)(unsigned)(uintptr_t)lB, 0, 0);
#else
      *(v8h*)lA = gA[c8];
      *(v8h*)lB = gB[c8];
#endif
    }
#if HAVE_ASYNC_LDS
    __builtin_amdgcn_s_wait_asynccnt(0);
#endif
    __syncthreads();

    // A fragment: lanes 0-15 row M=r K={0..7,16..23}; lanes 16-31 K={8..15,24..31}
    const v8h* ap = (const v8h*)&Ash[(wv * 16 + r) * LDK];
    v8h a0 = ap[hi];
    v8h a1 = ap[2 + hi];
    v16h afrag = __builtin_shufflevector(a0, a1,
        0,1,2,3,4,5,6,7,8,9,10,11,12,13,14,15);

#pragma unroll
    for (int t = 0; t < 4; ++t) {
      const v8h* bp = (const v8h*)&Bsh[(t * 16 + r) * LDK];
      v8h b0 = bp[hi];
      v8h b1 = bp[2 + hi];
      v16h bfrag = __builtin_shufflevector(b0, b1,
          0,1,2,3,4,5,6,7,8,9,10,11,12,13,14,15);
      acc[t] = __builtin_amdgcn_wmma_f32_16x16x32_f16(
          false, afrag, false, bfrag, (short)0, acc[t], false, false);
    }
    __syncthreads();
  }

  // C/D layout: VGPR q -> M = q (lanes 0-15) or q+8 (lanes 16-31); N = lane&15
#pragma unroll
  for (int t = 0; t < 4; ++t) {
#pragma unroll
    for (int q = 0; q < 8; ++q) {
      int row = bm + wv * 16 + q + hi * 8;
      int col = bn + t * 16 + r;
      float bb = bias ? bias[col] : 0.0f;
      C[(size_t)row * N + col] = acc[t][q] + bb;
    }
  }
}

// ---------------------------------------------------------------------------
// f32 -> f16 casts
// ---------------------------------------------------------------------------
__global__ void k_cast16(const float* __restrict__ X, _Float16* __restrict__ Y, int total8)
{
  int i = blockIdx.x * 256 + threadIdx.x;
  if (i >= total8) return;
  const float4* x4 = (const float4*)X;
  float4 a = x4[2 * i], b = x4[2 * i + 1];
  v8h o;
  o[0] = (_Float16)a.x; o[1] = (_Float16)a.y; o[2] = (_Float16)a.z; o[3] = (_Float16)a.w;
  o[4] = (_Float16)b.x; o[5] = (_Float16)b.y; o[6] = (_Float16)b.z; o[7] = (_Float16)b.w;
  *(v8h*)&Y[(size_t)i * 8] = o;
}

// cast + transpose weight [K,N] f32 -> [N,K] f16 (K is a power of two)
__global__ void k_wt16(const float* __restrict__ W, _Float16* __restrict__ Wt,
                       int N, int K, int kshift)
{
  int i = blockIdx.x * 256 + threadIdx.x;          // 8-element output groups
  if (i >= ((N * K) >> 3)) return;
  int k8 = (i << 3) & (K - 1);
  int n  = (i << 3) >> kshift;
  v8h o;
#pragma unroll
  for (int j = 0; j < 8; ++j) o[j] = (_Float16)W[(size_t)(k8 + j) * N + n];
  *(v8h*)&Wt[(size_t)n * K + k8] = o;
}

// ---------------------------------------------------------------------------
// Elementwise / small kernels
// ---------------------------------------------------------------------------
__global__ void k_c1(const float* __restrict__ x, const float* __restrict__ w,
                     const float* __restrict__ b, float* __restrict__ y, int BN)
{
  int i = blockIdx.x * 256 + threadIdx.x;
  if (i >= BN * 128) return;
  int tok = i >> 7, j = i & 127;
  y[i] = x[tok] * w[j] + b[j];
}

__global__ __launch_bounds__(128) void k_layernorm(
    const float* __restrict__ x, const float* __restrict__ w,
    const float* __restrict__ b, float* __restrict__ y, int T)
{
  int lane = threadIdx.x & 31;
  int t = blockIdx.x * 4 + (threadIdx.x >> 5);
  if (t >= T) return;
  const float4* xr = (const float4*)(x + (size_t)t * EMBED);
  float4 v0 = xr[lane], v1 = xr[lane + 32];
  float s = v0.x + v0.y + v0.z + v0.w + v1.x + v1.y + v1.z + v1.w;
#pragma unroll
  for (int o = 16; o > 0; o >>= 1) s += __shfl_xor(s, o, 32);
  float mu = s * (1.0f / EMBED);
  float vs = (v0.x-mu)*(v0.x-mu) + (v0.y-mu)*(v0.y-mu) + (v0.z-mu)*(v0.z-mu) + (v0.w-mu)*(v0.w-mu)
           + (v1.x-mu)*(v1.x-mu) + (v1.y-mu)*(v1.y-mu) + (v1.z-mu)*(v1.z-mu) + (v1.w-mu)*(v1.w-mu);
#pragma unroll
  for (int o = 16; o > 0; o >>= 1) vs += __shfl_xor(vs, o, 32);
  float inv = rsqrtf(vs * (1.0f / EMBED) + 1e-5f);
  const float4* w4 = (const float4*)w;
  const float4* b4 = (const float4*)b;
  float4* y4 = (float4*)(y + (size_t)t * EMBED);
  float4 wa = w4[lane], wb = w4[lane + 32], ba = b4[lane], bb = b4[lane + 32];
  float4 o0, o1;
  o0.x = (v0.x - mu) * inv * wa.x + ba.x;
  o0.y = (v0.y - mu) * inv * wa.y + ba.y;
  o0.z = (v0.z - mu) * inv * wa.z + ba.z;
  o0.w = (v0.w - mu) * inv * wa.w + ba.w;
  o1.x = (v1.x - mu) * inv * wb.x + bb.x;
  o1.y = (v1.y - mu) * inv * wb.y + bb.y;
  o1.z = (v1.z - mu) * inv * wb.z + bb.z;
  o1.w = (v1.w - mu) * inv * wb.w + bb.w;
  y4[lane] = o0;
  y4[lane + 32] = o1;
}

__global__ void k_pe_add(const float* __restrict__ coords,
                         const float* __restrict__ pw, const float* __restrict__ pb,
                         float* __restrict__ h, int BN)
{
  int i = blockIdx.x * 256 + threadIdx.x;
  if (i >= BN * EMBED) return;
  int tok = i >> 8, j = i & 255;
  const float* c = coords + (size_t)tok * 3;
  float v = c[0] * pw[j] + c[1] * pw[256 + j] + c[2] * pw[512 + j] + pb[j];
  h[i] += __sinf(30.0f * v);
}

// KNN: one thread per query, LDS tiles of 256 points, register top-16
// (unsorted -- softmax over the neighbor set is permutation invariant).
template<int C>
__global__ __launch_bounds__(256) void k_knn(
    const float* __restrict__ cbase, int cstride, int Npts, int* __restrict__ nn)
{
  __shared__ float tile[256 * C];
  int b = blockIdx.y;
  const float* cb = cbase + (size_t)b * Npts * cstride;
  int q = blockIdx.x * 256 + threadIdx.x;
  float qc[C];
#pragma unroll
  for (int c = 0; c < C; ++c) qc[c] = cb[(size_t)q * cstride + c];
  float best[KNN_K]; int bi[KNN_K];
#pragma unroll
  for (int j = 0; j < KNN_K; ++j) { best[j] = 3e38f; bi[j] = 0; }
  float wv = 3e38f; int ws = 0;
  for (int m0 = 0; m0 < Npts; m0 += 256) {
    __syncthreads();
    for (int e = threadIdx.x; e < 256 * C; e += 256) {
      int mm = e / C, cc = e % C;
      tile[cc * 256 + mm] = cb[(size_t)(m0 + mm) * cstride + cc];
    }
    __syncthreads();
    for (int mm = 0; mm < 256; ++mm) {
      float d = 0.f;
#pragma unroll
      for (int c = 0; c < C; ++c) { float df = qc[c] - tile[c * 256 + mm]; d += df * df; }
      if (d < wv) {
        int gm = m0 + mm;
#pragma unroll
        for (int j = 0; j < KNN_K; ++j) if (j == ws) { best[j] = d; bi[j] = gm; }
        wv = best[0]; ws = 0;
#pragma unroll
        for (int j = 1; j < KNN_K; ++j) if (best[j] > wv) { wv = best[j]; ws = j; }
      }
    }
  }
  int* out = nn + ((size_t)b * Npts + q) * KNN_K;
#pragma unroll
  for (int j = 0; j < KNN_K; ++j) out[j] = bi[j];
}

// neighbor attention: one thread per (batch, token, head); b128 q/k/v loads
__global__ __launch_bounds__(256) void k_attn(
    const float* __restrict__ qkv, const int* __restrict__ nn,
    float* __restrict__ out, int Btot, int Npts)
{
  int gid = blockIdx.x * 256 + threadIdx.x;
  if (gid >= Btot * HEADS) return;
  int hh = gid & 7;
  int tok = gid >> 3;
  int b = tok / Npts;
  const float4* qr4 = (const float4*)(qkv + (size_t)tok * 768 + hh * HD);
  const int* ir = nn + (size_t)tok * KNN_K;
  float q[HD];
#pragma unroll
  for (int t = 0; t < 8; ++t) {
    float4 f = qr4[t];
    q[4*t] = f.x; q[4*t+1] = f.y; q[4*t+2] = f.z; q[4*t+3] = f.w;
  }
  float s[KNN_K]; float mx = -3e38f;
#pragma unroll
  for (int j = 0; j < KNN_K; ++j) {
    int nbr = b * Npts + ir[j];
    const float4* kr4 = (const float4*)(qkv + (size_t)nbr * 768 + 256 + hh * HD);
    float a = 0.f;
#pragma unroll
    for (int t = 0; t < 8; ++t) {
      float4 f = kr4[t];
      a += q[4*t]*f.x + q[4*t+1]*f.y + q[4*t+2]*f.z + q[4*t+3]*f.w;
    }
    a *= 0.17677669529663687f;   // 1/sqrt(32)
    s[j] = a; mx = fmaxf(mx, a);
  }
  float se = 0.f;
#pragma unroll
  for (int j = 0; j < KNN_K; ++j) { s[j] = __expf(s[j] - mx); se += s[j]; }
  float inv = 1.0f / se;
  float o[HD];
#pragma unroll
  for (int d = 0; d < HD; ++d) o[d] = 0.f;
#pragma unroll
  for (int j = 0; j < KNN_K; ++j) {
    int nbr = b * Npts + ir[j];
    const float4* vr4 = (const float4*)(qkv + (size_t)nbr * 768 + 512 + hh * HD);
    float wj = s[j] * inv;
#pragma unroll
    for (int t = 0; t < 8; ++t) {
      float4 f = vr4[t];
      o[4*t] += wj*f.x; o[4*t+1] += wj*f.y; o[4*t+2] += wj*f.z; o[4*t+3] += wj*f.w;
    }
  }
  float4* o4 = (float4*)(out + (size_t)tok * EMBED + hh * HD);
#pragma unroll
  for (int t = 0; t < 8; ++t) {
    float4 f; f.x = o[4*t]; f.y = o[4*t+1]; f.z = o[4*t+2]; f.w = o[4*t+3];
    o4[t] = f;
  }
}

__global__ void k_metric(const float* __restrict__ qkv, float* __restrict__ met, int Btot)
{
  int i = blockIdx.x * 256 + threadIdx.x;
  if (i >= Btot * HD) return;
  int tok = i >> 5, d = i & 31;
  float s = 0.f;
#pragma unroll
  for (int hh = 0; hh < HEADS; ++hh) s += qkv[(size_t)tok * 768 + 256 + hh * HD + d];
  met[i] = s * (1.0f / HEADS);
}

__global__ void k_residual(float* __restrict__ h, const float* __restrict__ g,
                           const float* __restrict__ a, int total)
{
  int i = blockIdx.x * 256 + threadIdx.x;
  if (i < total) h[i] += g[i & (EMBED - 1)] * a[i];
}

// normalize metric rows, split even/odd token sets, emit f16 GEMM operands
__global__ void k_mnorm(const float* __restrict__ met, _Float16* __restrict__ ab,
                        _Float16* __restrict__ bb, int B_, int Npts)
{
  int tok = blockIdx.x * 256 + threadIdx.x;
  if (tok >= B_ * Npts) return;
  int b = tok / Npts, n = tok % Npts;
  const float* m = met + (size_t)tok * HD;
  float s = 0.f;
#pragma unroll
  for (int d = 0; d < HD; ++d) s += m[d] * m[d];
  float inv = 1.0f / (sqrtf(s) + 1e-6f);
  int nh = Npts >> 1;
  _Float16* dst = (n & 1) ? (bb + ((size_t)b * nh + (n >> 1)) * HD)
                          : (ab + ((size_t)b * nh + (n >> 1)) * HD);
#pragma unroll
  for (int d = 0; d < HD; ++d) dst[d] = (_Float16)(m[d] * inv);
}

// per-row max + argmax over scores (one wave per row)
__global__ __launch_bounds__(256) void k_rowmax(
    const float* __restrict__ sc, float* __restrict__ nmax,
    int* __restrict__ nidx, int totalRows, int Nb)
{
  int lane = threadIdx.x & 31;
  int r = blockIdx.x * 8 + (threadIdx.x >> 5);
  if (r >= totalRows) return;
  const float* row = sc + (size_t)r * Nb;
  float m = -3e38f; int mi = 0;
  for (int c = lane; c < Nb; c += 32) {
    float v = row[c];
    if (v > m) { m = v; mi = c; }
  }
#pragma unroll
  for (int o = 16; o > 0; o >>= 1) {
    float om = __shfl_xor(m, o, 32);
    int  oi = __shfl_xor(mi, o, 32);
    if (om > m || (om == m && oi < mi)) { m = om; mi = oi; }
  }
  if (lane == 0) { nmax[r] = m; nidx[r] = mi; }
}

// bitonic argsort (descending node_max): one block per batch, capacity 2048
__global__ __launch_bounds__(1024) void k_sort(
    const float* __restrict__ nmax, int* __restrict__ order, int Na)
{
  __shared__ float key[2048];
  __shared__ int   val[2048];
  int b = blockIdx.x;
  const float* km = nmax + (size_t)b * Na;
  for (int i = threadIdx.x; i < 2048; i += 1024) {
    key[i] = (i < Na) ? -km[i] : 3e38f;   // ascending of -x == descending of x
    val[i] = i;
  }
  __syncthreads();
  for (int ksz = 2; ksz <= 2048; ksz <<= 1) {
    for (int j = ksz >> 1; j > 0; j >>= 1) {
      for (int i = threadIdx.x; i < 2048; i += 1024) {
        int ixj = i ^ j;
        if (ixj > i) {
          bool up = ((i & ksz) == 0);
          float a = key[i], c = key[ixj];
          int va = val[i], vc = val[ixj];
          bool sw = up ? (a > c || (a == c && va > vc))
                       : (a < c || (a == c && va < vc));
          if (sw) { key[i] = c; key[ixj] = a; val[i] = vc; val[ixj] = va; }
        }
      }
      __syncthreads();
    }
  }
  for (int i = threadIdx.x; i < Na; i += 1024)
    order[(size_t)b * Na + i] = val[i];
}

__global__ void k_merge_init(const float* __restrict__ h, float* __restrict__ dst,
                             float* __restrict__ cnt, int B_, int Npts)
{
  int nh = Npts >> 1;
  int i = blockIdx.x * 256 + threadIdx.x;
  if (i >= B_ * nh * EMBED) return;
  int d = i & 255; int row = i >> 8;
  int b = row / nh; int j = row % nh;
  dst[i] = h[((size_t)b * Npts + 2 * j + 1) * EMBED + d];
  if (d == 0) cnt[row] = 1.0f;
}

__global__ void k_merge_scatter(const float* __restrict__ h, const int* __restrict__ order,
                                const int* __restrict__ nidx, float* __restrict__ dst,
                                float* __restrict__ cnt, int B_, int Npts)
{
  int nh = Npts >> 1;
  int i = blockIdx.x * 256 + threadIdx.x;
  if (i >= B_ * TOME_R * EMBED) return;
  int d = i & 255; int sr = i >> 8;
  int b = sr / TOME_R; int s = sr % TOME_R;
  int sa = order[(size_t)b * nh + s];
  int dj = nidx[(size_t)b * nh + sa];
  float v = h[((size_t)b * Npts + 2 * sa) * EMBED + d];
  atomicAdd(&dst[((size_t)b * nh + dj) * EMBED + d], v);
  if (d == 0) atomicAdd(&cnt[(size_t)b * nh + dj], 1.0f);
}

__global__ void k_merge_final(const float* __restrict__ h, const int* __restrict__ order,
                              const float* __restrict__ dst, const float* __restrict__ cnt,
                              float* __restrict__ hnew, int B_, int Npts)
{
  int nh = Npts >> 1;
  int nunm = nh - TOME_R;
  int Nout = Npts - TOME_R;
  int i = blockIdx.x * 256 + threadIdx.x;
  if (i >= B_ * Nout * EMBED) return;
  int d = i & 255; int row = i >> 8;
  int b = row / Nout; int t = row % Nout;
  float v;
  if (t < nunm) {
    int un = order[(size_t)b * nh + TOME_R + t];
    v = h[((size_t)b * Npts + 2 * un) * EMBED + d];
  } else {
    int j = t - nunm;
    v = dst[((size_t)b * nh + j) * EMBED + d] / cnt[(size_t)b * nh + j];
  }
  hnew[((size_t)b * Nout + t) * EMBED + d] = v;
}

__global__ void k_glu(const float* __restrict__ hid, float* __restrict__ g, int rows)
{
  int i = blockIdx.x * 256 + threadIdx.x;
  if (i >= rows * 1024) return;
  int r = i >> 10, j = i & 1023;
  float x1 = hid[(size_t)r * MLP_HID + j];
  float x2 = hid[(size_t)r * MLP_HID + 1024 + j];
  g[i] = (x1 / (1.0f + __expf(-x1))) * x2;   // silu(x1) * x2
}

// ---------------------------------------------------------------------------
// Host orchestration
// ---------------------------------------------------------------------------
extern "C" void kernel_launch(void* const* d_in, const int* in_sizes, int n_in,
                              void* d_out, int out_size, void* d_ws, size_t ws_size,
                              hipStream_t stream)
{
  (void)in_sizes; (void)n_in; (void)ws_size;
  const float* x      = (const float*)d_in[0];
  const float* coords = (const float*)d_in[1];
  auto P = [&](int i) { return (const float*)d_in[i]; };
  const float* c1_w = P(2);  const float* c1_b = P(3);
  const float* c2_w = P(4);  const float* c2_b = P(5);
  const float* tok_nw = P(6); const float* tok_nb = P(7);
  const float* pe_w = P(8);  const float* pe_b = P(9);

  const int B = 2, N0 = 4096;
  size_t off = 0;
  auto alloc_b = [&](size_t bytes) -> void* {
    void* p = (char*)d_ws + off;
    off += bytes;
    off = (off + 255) & ~(size_t)255;
    return p;
  };
  auto alloc_f = [&](size_t n) { return (float*)alloc_b(n * sizeof(float)); };
  auto alloc_h = [&](size_t n) { return (_Float16*)alloc_b(n * sizeof(_Float16)); };

  float* t1     = alloc_f((size_t)B * N0 * 128);
  float* h0     = alloc_f((size_t)B * N0 * EMBED);
  float* h1     = alloc_f((size_t)B * N0 * EMBED);
  float* xn     = alloc_f((size_t)B * N0 * EMBED);
  float* qkv    = alloc_f((size_t)B * N0 * 768);
  float* xa     = alloc_f((size_t)B * N0 * EMBED);
  float* tmpA   = alloc_f((size_t)B * N0 * EMBED);
  float* metric = alloc_f((size_t)B * N0 * HD);
  int*   nnidx  = (int*)alloc_b((size_t)B * N0 * KNN_K * sizeof(int));
  _Float16* a16 = alloc_h((size_t)B * 3584 * 1024);   // f16 activations (max: fc2 input)
  _Float16* w16 = alloc_h((size_t)MLP_HID * EMBED);   // f16 transposed weights (max: fc1)
  _Float16* ab16 = alloc_h((size_t)B * 2048 * HD);
  _Float16* bb16 = alloc_h((size_t)B * 2048 * HD);
  float* nmax   = alloc_f((size_t)B * 2048);
  int*   nidxb  = (int*)alloc_b((size_t)B * 2048 * sizeof(int));
  int*   order  = (int*)alloc_b((size_t)B * 2048 * sizeof(int));
  float* cnt    = alloc_f((size_t)B * 2048);
  float* dstb   = alloc_f((size_t)B * 2048 * EMBED);
  float* big    = alloc_f((size_t)B * 3584 * MLP_HID); // scores + mlp hidden (disjoint liveness)
  float* gated  = alloc_f((size_t)B * 3584 * 1024);

  auto gemm = [&](const float* Af32, const float* W, const float* bias, float* Cout,
                  int M, int N, int K, int kshift) {
    k_cast16<<<dim3(((M * K) / 8 + 255) / 256), 256, 0, stream>>>(Af32, a16, (M * K) / 8);
    k_wt16<<<dim3(((N * K) / 8 + 255) / 256), 256, 0, stream>>>(W, w16, N, K, kshift);
    k_gemm_wmma<<<dim3(N / 64, M / 64), 128, 0, stream>>>(a16, w16, bias, Cout, M, N, K);
  };

  // ---- token embedding: h = LN((x*c1)+b @ c2) + sin(30*(coords@pe)) ----
  {
    int BN = B * N0;
    k_c1<<<dim3((BN * 128 + 255) / 256), 256, 0, stream>>>(x, c1_w, c1_b, t1, BN);
    gemm(t1, c2_w, c2_b, xn, BN, EMBED, 128, 7);
    k_layernorm<<<dim3((BN + 3) / 4), 128, 0, stream>>>(xn, tok_nw, tok_nb, h0, BN);
    k_pe_add<<<dim3((BN * EMBED + 255) / 256), 256, 0, stream>>>(coords, pe_w, pe_b, h0, BN);
  }

  float* hcur = h0; float* hnxt = h1;
  int Ni = N0;
  for (int L = 0; L < DEPTH; ++L) {
    int pb = 10 + L * 14;
    const float *n1_w = P(pb+0), *n1_b = P(pb+1), *qkv_w = P(pb+2), *qkv_b = P(pb+3),
                *proj_w = P(pb+4), *proj_b = P(pb+5), *g1 = P(pb+6),
                *n2_w = P(pb+7), *n2_b = P(pb+8), *fc1_w = P(pb+9), *fc1_b = P(pb+10),
                *fc2_w = P(pb+11), *fc2_b = P(pb+12), *g2 = P(pb+13);
    int BN = B * Ni;

    // attention block
    k_layernorm<<<dim3((BN + 3) / 4), 128, 0, stream>>>(hcur, n1_w, n1_b, xn, BN);
    gemm(xn, qkv_w, qkv_b, qkv, BN, 768, EMBED, 8);
    if (L == 0)
      k_knn<3><<<dim3(Ni / 256, B), 256, 0, stream>>>(coords, 3, Ni, nnidx);
    else
      k_knn<LASTF><<<dim3(Ni / 256, B), 256, 0, stream>>>(
          hcur + (EMBED - LASTF), EMBED, Ni, nnidx);
    k_attn<<<dim3((BN * HEADS + 255) / 256), 256, 0, stream>>>(qkv, nnidx, xa, BN, Ni);
    k_metric<<<dim3((BN * HD + 255) / 256), 256, 0, stream>>>(qkv, metric, BN);
    gemm(xa, proj_w, proj_b, tmpA, BN, EMBED, EMBED, 8);
    k_residual<<<dim3((BN * EMBED + 255) / 256), 256, 0, stream>>>(
        hcur, g1, tmpA, BN * EMBED);

    // ToMe bipartite soft matching
    int nh = Ni >> 1;
    k_mnorm<<<dim3((BN + 255) / 256), 256, 0, stream>>>(metric, ab16, bb16, B, Ni);
    for (int b = 0; b < B; ++b)
      k_gemm_wmma<<<dim3(nh / 64, nh / 64), 128, 0, stream>>>(
          ab16 + (size_t)b * nh * HD, bb16 + (size_t)b * nh * HD, nullptr,
          big + (size_t)b * nh * nh, nh, nh, HD);
    k_rowmax<<<dim3((B * nh + 7) / 8), 256, 0, stream>>>(big, nmax, nidxb, B * nh, nh);
    k_sort<<<dim3(B), 1024, 0, stream>>>(nmax, order, nh);
    int Nout = Ni - TOME_R;
    k_merge_init<<<dim3((B * nh * EMBED + 255) / 256), 256, 0, stream>>>(
        hcur, dstb, cnt, B, Ni);
    k_merge_scatter<<<dim3((B * TOME_R * EMBED + 255) / 256), 256, 0, stream>>>(
        hcur, order, nidxb, dstb, cnt, B, Ni);
    k_merge_final<<<dim3((B * Nout * EMBED + 255) / 256), 256, 0, stream>>>(
        hcur, order, dstb, cnt, hnxt, B, Ni);

    // gated MLP on merged tokens
    int BM = B * Nout;
    k_layernorm<<<dim3((BM + 3) / 4), 128, 0, stream>>>(hnxt, n2_w, n2_b, xn, BM);
    gemm(xn, fc1_w, fc1_b, big, BM, MLP_HID, EMBED, 8);
    k_glu<<<dim3((BM * 1024 + 255) / 256), 256, 0, stream>>>(big, gated, BM);
    gemm(gated, fc2_w, fc2_b, tmpA, BM, EMBED, 1024, 10);
    k_residual<<<dim3((BM * EMBED + 255) / 256), 256, 0, stream>>>(
        hnxt, g2, tmpA, BM * EMBED);

    float* tsw = hcur; hcur = hnxt; hnxt = tsw;
    Ni = Nout;
  }

  (void)hipMemcpyAsync(d_out, hcur, (size_t)out_size * sizeof(float),
                       hipMemcpyDeviceToDevice, stream);
}
